// GriffinRecurrentBlock_61117384622325
// MI455X (gfx1250) — matrix operand order repaired and checked by reading, not compiled, not run on hardware
//
#include <hip/hip_runtime.h>
#include <cstdint>
#include <cstddef>

// ---------------------------------------------------------------------------
// Griffin recurrent block for MI455X (gfx1250), bf16 WMMA path.
//   B=4, T=2048, D=2048, L=2048, K(conv)=4
// 5 big GEMMs on v_wmma_f32_16x16x32_bf16, everything else f32.
// ---------------------------------------------------------------------------

constexpr int Bz = 4, Tz = 2048, Dz = 2048, Lz = 2048;
constexpr int Mrows = Bz * Tz;            // 8192
constexpr int BTL   = Bz * Tz * Lz;       // 16,777,216
constexpr int BTD   = Bz * Tz * Dz;       // 16,777,216

typedef __bf16 bf16_t;
typedef __attribute__((ext_vector_type(8)))  bf16_t bf16x8;
typedef __attribute__((ext_vector_type(16))) bf16_t bf16x16;
typedef __attribute__((ext_vector_type(8)))  float  v8f;

#define DEV static __device__ __forceinline__

DEV unsigned short f2bf(float f) {
  union { float f; unsigned u; } v; v.f = f;
  unsigned u = v.u;
  u += 0x7FFFu + ((u >> 16) & 1u);      // round-to-nearest-even
  return (unsigned short)(u >> 16);
}

// ---------------------------------------------------------------------------
// f32 -> bf16 elementwise convert
// ---------------------------------------------------------------------------
__global__ void k_f32_to_bf16(const float* __restrict__ in,
                              unsigned short* __restrict__ out, int n) {
  int i = blockIdx.x * blockDim.x + threadIdx.x;
  if (i < n) out[i] = f2bf(in[i]);
}

// ---------------------------------------------------------------------------
// f32 [R x C] -> bf16 [C x R] transpose+convert (weights), 32x32 LDS tile
// ---------------------------------------------------------------------------
__global__ void k_transpose_bf16(const float* __restrict__ in,
                                 unsigned short* __restrict__ out,
                                 int R, int C) {
  __shared__ float tile[32][33];
  const int bx = blockIdx.x * 32;   // col base in `in`
  const int by = blockIdx.y * 32;   // row base in `in`
  const int tx = threadIdx.x;       // 0..31
  const int ty = threadIdx.y;       // 0..7
  #pragma unroll
  for (int r = ty; r < 32; r += 8)
    tile[r][tx] = in[(size_t)(by + r) * C + (bx + tx)];
  __syncthreads();
  #pragma unroll
  for (int r = ty; r < 32; r += 8)
    out[(size_t)(bx + r) * R + (by + tx)] = f2bf(tile[tx][r]);
}

// ---------------------------------------------------------------------------
// bf16 WMMA GEMM:  C[M,N] = act( A[M,K] * BT[N,K]^T + bias[N] )
// block = 256 threads (8 waves), block tile 128(M) x 256(N),
// wave tile 64x64 = 4x4 tiles of v_wmma_f32_16x16x32_bf16, K step 32.
// M,N,K all multiples of the tiling (8192 / 2048 / 2048 here).
// ---------------------------------------------------------------------------
enum { ACT_NONE = 0, ACT_GELU = 1, ACT_SIG = 2 };

template <int ACT>
__global__ __launch_bounds__(256) void k_gemm_bf16(
    const unsigned short* __restrict__ A,    // M x K row-major (bf16 bits)
    const unsigned short* __restrict__ BT,   // N x K row-major (bf16 bits)
    const float* __restrict__ bias,          // N
    float* __restrict__ C,                   // M x N f32
    int M, int N, int K) {
  (void)M;
  const int lane = threadIdx.x & 31;
  const int wave = threadIdx.x >> 5;          // 0..7
  const int hh   = lane >> 4;                 // half-wave: 0 or 1
  const int l16  = lane & 15;

  const int waveM = blockIdx.y * 128 + (wave >> 2) * 64;
  const int waveN = blockIdx.x * 256 + (wave & 3) * 64;

  v8f acc[4][4];
  const v8f vzero = {0.f, 0.f, 0.f, 0.f, 0.f, 0.f, 0.f, 0.f};
  #pragma unroll
  for (int i = 0; i < 4; ++i)
    #pragma unroll
    for (int j = 0; j < 4; ++j) acc[i][j] = vzero;

  for (int k0 = 0; k0 < K; k0 += 32) {
    // A fragments (contiguous k order, same as B):
    //   lane half 0: elems 0..15 = K k0+0..15
    //   lane half 1: elems 0..15 = K k0+16..31
    bf16x16 afr[4], bfr[4];
    #pragma unroll
    for (int i = 0; i < 4; ++i) {
      const unsigned short* ap =
          A + (size_t)(waveM + i * 16 + l16) * K + k0 + hh * 16;
      bf16x8 c0 = *(const bf16x8*)(ap);
      bf16x8 c1 = *(const bf16x8*)(ap + 8);
      afr[i] = __builtin_shufflevector(c0, c1, 0, 1, 2, 3, 4, 5, 6, 7,
                                       8, 9, 10, 11, 12, 13, 14, 15);
    }
    // B fragments: lane holds column n = lane%16, contiguous 16 K values
    // starting at k0 + (lane/16)*16  (BT is N x K row-major).
    #pragma unroll
    for (int j = 0; j < 4; ++j) {
      const unsigned short* bp =
          BT + (size_t)(waveN + j * 16 + l16) * K + k0 + hh * 16;
      bfr[j] = *(const bf16x16*)bp;
    }
    #pragma unroll
    for (int i = 0; i < 4; ++i)
      #pragma unroll
      for (int j = 0; j < 4; ++j)
        acc[i][j] = __builtin_amdgcn_wmma_f32_16x16x32_bf16(
            false, afr[i], false, bfr[j], (short)0, acc[i][j], false, false);
  }

  // Epilogue: D layout — lane half h, VGPR r -> row = h*8 + r, col = lane%16
  #pragma unroll
  for (int i = 0; i < 4; ++i) {
    #pragma unroll
    for (int j = 0; j < 4; ++j) {
      const int col = waveN + j * 16 + l16;
      const float bv = bias[col];
      #pragma unroll
      for (int r = 0; r < 8; ++r) {
        const int row = waveM + i * 16 + hh * 8 + r;
        float v = acc[i][j][r] + bv;
        if (ACT == ACT_GELU)
          v = 0.5f * v * (1.0f + erff(v * 0.70710678118654752f));
        else if (ACT == ACT_SIG)
          v = 1.0f / (1.0f + expf(-v));
        C[(size_t)row * N + col] = v;
      }
    }
  }
}

// ---------------------------------------------------------------------------
// Causal depthwise conv1d (width 4), in place over xb (per-column rolling
// window -> no cross-block hazard). Also emits bf16 copy for GEMM2 input.
// ---------------------------------------------------------------------------
__global__ void k_conv(float* __restrict__ xb, unsigned short* __restrict__ cbf,
                       const float* __restrict__ conv_w,
                       const float* __restrict__ conv_b) {
  int tid = blockIdx.x * blockDim.x + threadIdx.x;   // 0 .. B*L-1
  if (tid >= Bz * Lz) return;
  const int b = tid / Lz, l = tid - b * Lz;
  const float w0 = conv_w[0 * Lz + l], w1 = conv_w[1 * Lz + l];
  const float w2 = conv_w[2 * Lz + l], w3 = conv_w[3 * Lz + l];
  const float cb = conv_b[l];
  float xm3 = 0.f, xm2 = 0.f, xm1 = 0.f;
  const size_t base = (size_t)b * Tz * Lz + l;
  for (int t = 0; t < Tz; ++t) {
    const size_t idx = base + (size_t)t * Lz;
    const float cur = xb[idx];
    const float v = cb + w0 * xm3 + w1 * xm2 + w2 * xm1 + w3 * cur;
    xb[idx] = v;
    cbf[idx] = f2bf(v);
    xm3 = xm2; xm2 = xm1; xm1 = cur;
  }
}

// ---------------------------------------------------------------------------
// RG-LRU gates: in gx = sigmoid(i-gate), ga = sigmoid(a-gate), conv (f32).
// Writes a -> ga buffer, nx -> gx buffer (element-wise in place).
// ---------------------------------------------------------------------------
__global__ void k_rglru(float* __restrict__ gx, float* __restrict__ ga,
                        const float* __restrict__ conv,
                        const float* __restrict__ a_param, int n) {
  int i = blockIdx.x * blockDim.x + threadIdx.x;
  if (i >= n) return;
  const int l = i % Lz;
  const int t = (i / Lz) % Tz;
  const float sp = log1pf(expf(a_param[l]));      // softplus
  const float la = -8.0f * ga[i] * sp;
  float a = expf(la);
  float mult = sqrtf(fmaxf(0.f, 1.0f - expf(2.0f * la)));
  if (t == 0) { a = 0.f; mult = 1.f; }
  const float nx = gx[i] * conv[i] * mult;
  ga[i] = a;
  gx[i] = nx;
}

// ---------------------------------------------------------------------------
// Diagonal linear scan h_t = a_t*h_{t-1} + nx_t ; z = h*y (bf16) ; hn out.
// One thread per (b,l); lanes cover consecutive l -> coalesced per t-step.
// ---------------------------------------------------------------------------
__global__ void k_scan(const float* __restrict__ a, const float* __restrict__ nx,
                       const float* __restrict__ y,
                       unsigned short* __restrict__ z_bf16,
                       float* __restrict__ hn) {
  int tid = blockIdx.x * blockDim.x + threadIdx.x;
  if (tid >= Bz * Lz) return;
  const int b = tid / Lz, l = tid - b * Lz;
  const size_t base = (size_t)b * Tz * Lz + l;
  float h = 0.f;
  for (int t = 0; t < Tz; ++t) {
    const size_t idx = base + (size_t)t * Lz;
    h = a[idx] * h + nx[idx];
    z_bf16[idx] = f2bf(h * y[idx]);
  }
  hn[tid] = h;
}

// ---------------------------------------------------------------------------
extern "C" void kernel_launch(void* const* d_in, const int* in_sizes, int n_in,
                              void* d_out, int out_size, void* d_ws,
                              size_t ws_size, hipStream_t stream) {
  (void)in_sizes; (void)n_in; (void)out_size; (void)ws_size;
  const float* x       = (const float*)d_in[0];
  const float* w_y     = (const float*)d_in[1];
  const float* b_y     = (const float*)d_in[2];
  const float* w_x     = (const float*)d_in[3];
  const float* b_x     = (const float*)d_in[4];
  const float* conv_w  = (const float*)d_in[5];
  const float* conv_b  = (const float*)d_in[6];
  const float* a_param = (const float*)d_in[7];
  const float* w_ig    = (const float*)d_in[8];
  const float* b_ig    = (const float*)d_in[9];
  const float* w_ag    = (const float*)d_in[10];
  const float* b_ag    = (const float*)d_in[11];
  const float* w_out   = (const float*)d_in[12];
  const float* b_out   = (const float*)d_in[13];

  float* out = (float*)d_out;                 // [B,T,D]
  float* hn  = out + (size_t)BTD;             // [B,L] appended

  // ---- workspace carve-out (~411 MB) ----
  char* ws = (char*)d_ws;
  size_t off = 0;
  auto carve = [&](size_t bytes) -> void* {
    void* p = ws + off;
    off = (off + bytes + 255) & ~(size_t)255;
    return p;
  };
  unsigned short* x_bf  = (unsigned short*)carve((size_t)BTD * 2);
  unsigned short* wyT   = (unsigned short*)carve((size_t)Dz * Lz * 2);
  unsigned short* wxT   = (unsigned short*)carve((size_t)Dz * Lz * 2);
  unsigned short* wigT  = (unsigned short*)carve((size_t)Lz * Lz * 2);
  unsigned short* wagT  = (unsigned short*)carve((size_t)Lz * Lz * 2);
  unsigned short* woutT = (unsigned short*)carve((size_t)Lz * Dz * 2);
  float*          ybuf  = (float*)carve((size_t)BTL * 4);
  float*          xbbuf = (float*)carve((size_t)BTL * 4);   // xb, then conv (in place)
  unsigned short* cbf   = (unsigned short*)carve((size_t)BTL * 2);
  float*          gxbuf = (float*)carve((size_t)BTL * 4);   // sigmoid(ix) -> nx
  float*          gabuf = (float*)carve((size_t)BTL * 4);   // sigmoid(ax) -> a
  unsigned short* zbf   = (unsigned short*)carve((size_t)BTL * 2);

  // ---- 1) precision staging ----
  k_f32_to_bf16<<<BTD / 256, 256, 0, stream>>>(x, x_bf, BTD);
  dim3 tb(32, 8), tg(Lz / 32, Dz / 32);
  k_transpose_bf16<<<tg, tb, 0, stream>>>(w_y,   wyT,   Dz, Lz); // -> [L,D]
  k_transpose_bf16<<<tg, tb, 0, stream>>>(w_x,   wxT,   Dz, Lz); // -> [L,D]
  k_transpose_bf16<<<tg, tb, 0, stream>>>(w_ig,  wigT,  Lz, Lz); // -> [L,L]
  k_transpose_bf16<<<tg, tb, 0, stream>>>(w_ag,  wagT,  Lz, Lz); // -> [L,L]
  k_transpose_bf16<<<tg, tb, 0, stream>>>(w_out, woutT, Lz, Dz); // -> [D,L]

  // ---- 2) projections: y = GELU(x@w_y+b), xb = x@w_x+b ----
  dim3 gg(Lz / 256, Mrows / 128);
  k_gemm_bf16<ACT_GELU><<<gg, 256, 0, stream>>>(x_bf, wyT, b_y, ybuf,
                                                Mrows, Lz, Dz);
  k_gemm_bf16<ACT_NONE><<<gg, 256, 0, stream>>>(x_bf, wxT, b_x, xbbuf,
                                                Mrows, Lz, Dz);

  // ---- 3) causal depthwise conv (in place) ----
  k_conv<<<(Bz * Lz) / 256, 256, 0, stream>>>(xbbuf, cbf, conv_w, conv_b);

  // ---- 4) gates: sigmoid(conv@w_ig+b), sigmoid(conv@w_ag+b) ----
  k_gemm_bf16<ACT_SIG><<<gg, 256, 0, stream>>>(cbf, wigT, b_ig, gxbuf,
                                               Mrows, Lz, Lz);
  k_gemm_bf16<ACT_SIG><<<gg, 256, 0, stream>>>(cbf, wagT, b_ag, gabuf,
                                               Mrows, Lz, Lz);

  // ---- 5) RG-LRU elementwise, then diagonal scan ----
  k_rglru<<<BTL / 256, 256, 0, stream>>>(gxbuf, gabuf, xbbuf, a_param, BTL);
  k_scan<<<(Bz * Lz) / 256, 256, 0, stream>>>(gabuf, gxbuf, ybuf, zbf, hn);

  // ---- 6) output projection ----
  dim3 gg3(Dz / 256, Mrows / 128);
  k_gemm_bf16<ACT_NONE><<<gg3, 256, 0, stream>>>(zbf, woutT, b_out, out,
                                                 Mrows, Dz, Lz);
}
